// Audio_Visual_aug_model_39762807226727
// MI455X (gfx1250) — matrix operand order repaired and checked by reading, not compiled, not run on hardware
//
#include <hip/hip_runtime.h>
#include <hip/hip_bf16.h>

typedef __attribute__((ext_vector_type(16))) _Float16 v16h;
typedef __attribute__((ext_vector_type(8)))  _Float16 v8h_t;
typedef __attribute__((ext_vector_type(8)))  float    v8f;

union V16U { v16h v; v8h_t h[2]; };

#define SH 528   // LDS half-stride for a 48x512 tile (16-half pad, keeps 16B align)
#define TS 52    // float stride for 48x48 tanh buffer
#define VTILE (48 * SH)          // halves per V tile
#define SMEM_BYTES (2 * VTILE * 2 + 2 * 48 * TS * 4 + (96 + 48 + 96 + 96) * 4)

// ---------------------------------------------------------------------------
// Prep 1: convert W_video (512x512) and W_v (48x512) fp32 -> f16 into ws
// ---------------------------------------------------------------------------
__global__ void av_cvt_weights(const float* __restrict__ Wvideo,
                               const float* __restrict__ Wv,
                               _Float16* __restrict__ o) {
  const int NW = 512 * 512;
  int i = blockIdx.x * 256 + threadIdx.x;
  if (i < NW)                 o[i] = (_Float16)Wvideo[i];
  else if (i < NW + 48 * 512) o[i] = (_Float16)Wv[i - NW];
}

// ---------------------------------------------------------------------------
// Prep 2: per (b,t): a = relu(audio @ W_audio^T + b_audio); inter = a @ W_g^T
// ---------------------------------------------------------------------------
__global__ __launch_bounds__(256) void av_audio_inter(
    const float* __restrict__ audio, const float* __restrict__ W_audio,
    const float* __restrict__ b_audio, const float* __restrict__ W_g,
    float* __restrict__ inter) {
  __shared__ float au[128];
  __shared__ float as_[512];
  const int bt = blockIdx.x, tid = threadIdx.x;
  if (tid < 128) au[tid] = audio[(size_t)bt * 128 + tid];
  __syncthreads();
  for (int h = tid; h < 512; h += 256) {
    float s = b_audio[h];
    const float* w = W_audio + (size_t)h * 128;
#pragma unroll 8
    for (int k = 0; k < 128; ++k) s += au[k] * w[k];
    as_[h] = s > 0.f ? s : 0.f;
  }
  __syncthreads();
  if (tid < 48) {
    float s = 0.f;
    const float* wg = W_g + (size_t)tid * 512;
#pragma unroll 8
    for (int h = 0; h < 512; ++h) s += wg[h] * as_[h];
    inter[(size_t)bt * 48 + tid] = s;
  }
}

// ---------------------------------------------------------------------------
// Main fused kernel: TWO (b,t) items per block (amortizes W_video B-fragment
// L2 traffic over 6 M-tiles). 256 threads = 8 waves (wave32).
// ---------------------------------------------------------------------------
__global__ __launch_bounds__(256, 1) void av_main(
    const float* __restrict__ video, const float* __restrict__ b_video,
    const _Float16* __restrict__ Wv16, const _Float16* __restrict__ Wvv16,
    const float* __restrict__ inter, const float* __restrict__ W_h,
    float* __restrict__ out) {
  extern __shared__ char smem[];
  _Float16* Vh   = (_Float16*)smem;                       // 2 V tiles (reused for v)
  float* tbuf    = (float*)(smem + 2 * VTILE * 2);        // 2 * 48*TS
  float* inter_s = tbuf + 2 * 48 * TS;                    // 2*48
  float* wh_s    = inter_s + 96;                          // 48
  float* zbuf    = wh_s + 48;                             // 2*48
  float* alpha_s = zbuf + 96;                             // 2*48

  const int bt0 = blockIdx.x * 2;
  const int tid = threadIdx.x;
  const int wv = tid >> 5;               // wave id 0..7
  const int l  = tid & 31;               // lane in wave
  const int lr = l & 15;                 // row/col within half-wave
  const int lh = l >> 4;                 // 0 = lanes 0-15, 1 = lanes 16-31

  const float* Vg0 = video + (size_t)bt0 * (48 * 512);

  if (tid < 96) inter_s[tid] = inter[(size_t)bt0 * 48 + tid];
  if (tid < 48) wh_s[tid] = W_h[tid];

  // ---- Phase 1: stage both V tiles (48x512 fp32 each) into LDS as f16 ----
  for (int i = tid; i < 2 * (48 * 512) / 8; i += 256) {
    const int base = i * 8;                    // flat over both tiles
    const int ib   = base >> 14;               // /(48*512) since 24576=2^14+2^13 -> careful
    const int rem  = base - ib * 24576;
    const int row  = rem >> 9;
    const int col  = rem & 511;
    const float4* p = (const float4*)(Vg0 + base);
    float4 x = p[0], y = p[1];
    v8h_t h;
    h[0] = (_Float16)x.x; h[1] = (_Float16)x.y;
    h[2] = (_Float16)x.z; h[3] = (_Float16)x.w;
    h[4] = (_Float16)y.x; h[5] = (_Float16)y.y;
    h[6] = (_Float16)y.z; h[7] = (_Float16)y.w;
    *(v8h_t*)&Vh[ib * VTILE + row * SH + col] = h;
  }
  __syncthreads();

  // ---- Phase 2: GEMM1 for both items: v = relu(V @ W_video^T + b) --------
  // Per wave: 2 items x 3 M-tiles x 4 N-tiles, K=512 in 16 steps of 32.
  v8f acc[2][3][4];
#pragma unroll
  for (int ib = 0; ib < 2; ++ib)
#pragma unroll
    for (int mt = 0; mt < 3; ++mt)
#pragma unroll
      for (int nt = 0; nt < 4; ++nt) acc[ib][mt][nt] = 0.f;

  const int n0 = wv * 64;
  for (int ks = 0; ks < 16; ++ks) {
    const int k0 = ks * 32;
    // B fragments: loaded ONCE, used by both items (6 M-tiles total)
    v16h Bf[4];
#pragma unroll
    for (int nt = 0; nt < 4; ++nt) {
      // 16-bit B, k order used here (as A): lanes 0-15 hold K=0..15 of column n, lanes 16-31 K=16..31
      const _Float16* p = Wv16 + (size_t)(n0 + nt * 16 + lr) * 512 + k0 + lh * 16;
      Bf[nt] = *(const v16h*)p;
    }
#pragma unroll
    for (int ib = 0; ib < 2; ++ib) {
      v16h A[3];
#pragma unroll
      for (int mt = 0; mt < 3; ++mt) {
        // 16-bit A, k order used here (as B): lanes 0-15 hold K 0..15, lanes 16-31 K 16..31
        const _Float16* p = &Vh[ib * VTILE + (mt * 16 + lr) * SH + k0 + lh * 16];
        V16U u;
        u.h[0] = *(const v8h_t*)p;
        u.h[1] = *(const v8h_t*)(p + 8);
        A[mt] = u.v;
      }
#pragma unroll
      for (int mt = 0; mt < 3; ++mt)
#pragma unroll
        for (int nt = 0; nt < 4; ++nt)
          acc[ib][mt][nt] = __builtin_amdgcn_wmma_f32_16x16x32_f16(
              false, A[mt], false, Bf[nt], (short)0, acc[ib][mt][nt], false, false);
    }
  }
  __syncthreads();  // all waves done reading Vh; results live in VGPRs

  // store relu(acc + bias) as f16 back into the SAME LDS buffers ("vh")
#pragma unroll
  for (int nt = 0; nt < 4; ++nt) {
    const int col = n0 + nt * 16 + lr;
    const float bn = b_video[col];
#pragma unroll
    for (int ib = 0; ib < 2; ++ib)
#pragma unroll
      for (int mt = 0; mt < 3; ++mt)
#pragma unroll
        for (int r = 0; r < 8; ++r) {
          const int row = mt * 16 + r + lh * 8;   // C/D layout: VGPR r -> M=r / r+8
          float x = acc[ib][mt][nt][r] + bn;
          Vh[ib * VTILE + row * SH + col] = (_Float16)(x > 0.f ? x : 0.f);
        }
  }
  __syncthreads();

  // ---- Phase 3: GEMM2 content = v @ W_v^T (48x48, K=512), waves 0..5 -----
  if (wv < 6) {
    const int ib = wv / 3;
    const int mt = wv % 3;
    v8f c2[3];
#pragma unroll
    for (int nt = 0; nt < 3; ++nt) c2[nt] = 0.f;
    for (int ks = 0; ks < 16; ++ks) {
      const int k0 = ks * 32;
      const _Float16* p = &Vh[ib * VTILE + (mt * 16 + lr) * SH + k0 + lh * 16];
      V16U u;
      u.h[0] = *(const v8h_t*)p;
      u.h[1] = *(const v8h_t*)(p + 8);
      v16h A2 = u.v;
#pragma unroll
      for (int nt = 0; nt < 3; ++nt) {
        const _Float16* q = Wvv16 + (size_t)(nt * 16 + lr) * 512 + k0 + lh * 16;
        v16h B2 = *(const v16h*)q;
        c2[nt] = __builtin_amdgcn_wmma_f32_16x16x32_f16(
            false, A2, false, B2, (short)0, c2[nt], false, false);
      }
    }
#pragma unroll
    for (int nt = 0; nt < 3; ++nt)
#pragma unroll
      for (int r = 0; r < 8; ++r) {
        const int row = mt * 16 + r + lh * 8;     // n index
        const int col = nt * 16 + lr;             // m index
        const float t = tanhf(c2[nt][r] + inter_s[ib * 48 + row]) * wh_s[col];
        tbuf[ib * 48 * TS + row * TS + col] = t;
      }
  }
  __syncthreads();

  // ---- z = row-sum, softmax over N=48 (both items) -----------------------
  if (tid < 96) {
    const int ib = tid / 48, n = tid % 48;
    float s = 0.f;
#pragma unroll 4
    for (int m = 0; m < 48; ++m) s += tbuf[ib * 48 * TS + n * TS + m];
    zbuf[tid] = s;
  }
  __syncthreads();
  if (tid < 2) {
    const int ib = tid;
    float mx = zbuf[ib * 48];
    for (int n = 1; n < 48; ++n) mx = fmaxf(mx, zbuf[ib * 48 + n]);
    float sum = 0.f;
    for (int n = 0; n < 48; ++n) sum += expf(zbuf[ib * 48 + n] - mx);
    const float inv = 1.f / sum;
    for (int n = 0; n < 48; ++n)
      alpha_s[ib * 48 + n] = expf(zbuf[ib * 48 + n] - mx) * inv;
  }
  __syncthreads();

  // ---- Phase 4: c[h] = sum_n alpha[n] * V[n][h]  (fp32 from global/L2) ---
#pragma unroll
  for (int ib = 0; ib < 2; ++ib) {
    const float* Vg = Vg0 + (size_t)ib * (48 * 512);
    for (int h = tid; h < 512; h += 256) {
      float s = 0.f;
#pragma unroll 8
      for (int n = 0; n < 48; ++n) s += alpha_s[ib * 48 + n] * Vg[n * 512 + h];
      out[(size_t)(bt0 + ib) * 512 + h] = s;
    }
  }
}

// ---------------------------------------------------------------------------
extern "C" void kernel_launch(void* const* d_in, const int* in_sizes, int n_in,
                              void* d_out, int out_size, void* d_ws, size_t ws_size,
                              hipStream_t stream) {
  const float* audio   = (const float*)d_in[0];  // [2048,128]
  const float* video   = (const float*)d_in[1];  // [2048,48,512]
  const float* W_audio = (const float*)d_in[2];  // [512,128]
  const float* b_audio = (const float*)d_in[3];  // [512]
  const float* W_video = (const float*)d_in[4];  // [512,512]
  const float* b_video = (const float*)d_in[5];  // [512]
  const float* W_v     = (const float*)d_in[6];  // [48,512]
  const float* W_g     = (const float*)d_in[7];  // [48,512]
  const float* W_h     = (const float*)d_in[8];  // [1,48]
  float* out = (float*)d_out;

  // workspace layout
  _Float16* Wv16  = (_Float16*)d_ws;                       // 512*512 halves
  _Float16* Wvv16 = Wv16 + 512 * 512;                      // 48*512 halves
  float* inter = (float*)((char*)d_ws + (size_t)(512 * 512 + 48 * 512) * 2); // 2048*48 f32

  const int BT = 2048;

  (void)hipFuncSetAttribute((const void*)av_main,
                            hipFuncAttributeMaxDynamicSharedMemorySize,
                            SMEM_BYTES);

  av_cvt_weights<<<(512 * 512 + 48 * 512 + 255) / 256, 256, 0, stream>>>(
      W_video, W_v, Wv16);
  av_audio_inter<<<BT, 256, 0, stream>>>(audio, W_audio, b_audio, W_g, inter);
  av_main<<<BT / 2, 256, SMEM_BYTES, stream>>>(video, b_video, Wv16, Wvv16,
                                               inter, W_h, out);
}
